// GRU_7292854469343
// MI455X (gfx1250) — matrix operand order repaired and checked by reading, not compiled, not run on hardware
//
#include <hip/hip_runtime.h>
#include <hip/hip_bf16.h>

typedef __attribute__((ext_vector_type(16))) _Float16 v16h;
typedef __attribute__((ext_vector_type(8)))  _Float16 v8h;
typedef __attribute__((ext_vector_type(4)))  _Float16 v4h;
typedef __attribute__((ext_vector_type(8)))  float    v8f;
typedef int v4i __attribute__((vector_size(16)));   // matches builtin's expected pointee

constexpr int SEQ  = 2048;
constexpr int BAT  = 64;
constexpr int HID  = 512;
constexpr int NBLK = 16;          // persistent workgroups, each owns 32 hidden outputs
constexpr int NSL  = HID / NBLK;  // 32
constexpr int LDW  = HID + 8;     // LDS row pitch (halves) for weights, 16B aligned
constexpr int LDH  = HID + 8;     // LDS row pitch (halves) for h / r*h

#if __has_builtin(__builtin_amdgcn_global_load_async_to_lds_b128)
#define GRU_ASYNC_LDS 1
#endif

// ---------------- fragment helpers (wave32 WMMA layouts, ISA 7.12.2) ----------

__device__ __forceinline__ v16h make_frag(v8h lo, v8h hi) {
  v16h r;
#pragma unroll
  for (int i = 0; i < 8; ++i) { r[i] = lo[i]; r[i + 8] = hi[i]; }
  return r;
}

// A (16x32, M x K) from row-major [rows][ld] array.
__device__ __forceinline__ v16h load_a(const _Float16* base, int ld, int m0, int k0, int lane) {
  const _Float16* p = base + (m0 + (lane & 15)) * ld + k0 + ((lane >> 4) << 3);
  v8h lo = *(const v8h*)p;
  v8h hi = *(const v8h*)(p + 16);
  return make_frag(lo, hi);
}

// B (32x16, K x N) from column-major storage [cols][ldk] (each column contiguous in K).
__device__ __forceinline__ v16h load_b(const _Float16* base, int ldk, int k0, int n0, int lane) {
  const _Float16* p = base + (n0 + (lane & 15)) * ldk + k0 + ((lane >> 4) << 3);
  v8h lo = *(const v8h*)p;
  v8h hi = *(const v8h*)(p + 16);
  return make_frag(lo, hi);
}

__device__ __forceinline__ v8f wmma_f16(v16h a, v16h b, v8f c) {
  return __builtin_amdgcn_wmma_f32_16x16x32_f16(false, a, false, b, (short)0, c, false, false);
}

// ---- global[64*512 f16] -> LDS[64][LDH] copy (async-to-LDS path when available)

__device__ __forceinline__ void wait_async_lds() {
#ifdef GRU_ASYNC_LDS
#if __has_builtin(__builtin_amdgcn_s_wait_asynccnt)
  __builtin_amdgcn_s_wait_asynccnt(0);
#else
  asm volatile("s_wait_asynccnt 0x0" ::: "memory");
#endif
#endif
}

__device__ __forceinline__ void copy_h_to_lds(_Float16* lds_dst, const _Float16* gsrc, int tid) {
#ifdef GRU_ASYNC_LDS
#pragma unroll 4
  for (int it = 0; it < (BAT * HID) / (256 * 8); ++it) {  // 16 passes, 16B/lane
    int idx = (it * 256 + tid) * 8;
    int r = idx >> 9, c = idx & (HID - 1);
    __builtin_amdgcn_global_load_async_to_lds_b128(
        (__attribute__((address_space(1))) v4i*)&gsrc[idx],
        (__attribute__((address_space(3))) v4i*)&lds_dst[r * LDH + c],
        0, 0);
  }
#else
  for (int idx = tid * 8; idx < BAT * HID; idx += 256 * 8) {
    int r = idx >> 9, c = idx & (HID - 1);
    *(v8h*)&lds_dst[r * LDH + c] = *(const v8h*)&gsrc[idx];
  }
#endif
}

// ---------------- kernel 1: f32 weights -> f16, split recurrent/input halves ----

__global__ __launch_bounds__(256) void gru_prep(
    const float* __restrict__ Wz, const float* __restrict__ Wr, const float* __restrict__ Wh,
    _Float16* __restrict__ Wrec,  // [3][HID n][HID k]  (k < HID half of W)
    _Float16* __restrict__ Wx)    // [3][HID n][HID k]  (k >= HID half of W)
{
  int idx = blockIdx.x * 256 + threadIdx.x;  // over 3*512*1024
  if (idx >= 3 * HID * 2 * HID) return;
  int g = idx / (HID * 2 * HID);
  int r = idx - g * (HID * 2 * HID);
  int n = r >> 10;
  int k = r & 1023;
  const float* W = (g == 0) ? Wz : (g == 1) ? Wr : Wh;
  float v = W[n * (2 * HID) + k];
  if (k < HID) Wrec[((size_t)g * HID + n) * HID + k] = (_Float16)v;
  else         Wx  [((size_t)g * HID + n) * HID + (k - HID)] = (_Float16)v;
}

// ---------------- kernel 2: batched input-side gate GEMM (fully parallel) ------
// Xg[g][t*64+b][n] = b_g[n] + emb[x[b][t]] . Wx[g][n][:]

__global__ __launch_bounds__(256) void gru_xgates(
    const int* __restrict__ x, const float* __restrict__ emb,
    const _Float16* __restrict__ Wx,
    const float* __restrict__ bz, const float* __restrict__ br, const float* __restrict__ bh,
    _Float16* __restrict__ Xg)
{
  __shared__ _Float16 Ash[64][40];  // staged embedding rows, f16, padded pitch
  __shared__ int tok[64];

  const int tid = threadIdx.x, lane = tid & 31, wave = tid >> 5;
  const int g  = blockIdx.z;            // gate
  const int Mt = blockIdx.y;            // 64-row tile of flat (t*64+b)
  const int Nt = blockIdx.x;            // 64-col tile of hidden
  const int m0 = (wave & 3) << 4;
  const int n0 = Nt * 64 + ((wave >> 2) << 5);  // this wave handles 2 n-tiles

  if (tid < 64) {
    int i = Mt * 64 + tid;              // flat row: t = i>>6, b = i&63
    tok[tid] = x[(i & 63) * SEQ + (i >> 6)];
  }
  __syncthreads();

  const _Float16* Wg = Wx + (size_t)g * HID * HID;
  v8f acc0 = {}, acc1 = {};
  for (int k0 = 0; k0 < HID; k0 += 32) {
#pragma unroll
    for (int e = 0; e < 2; ++e) {       // stage 64x32 A slice: float4 load -> 4xf16 store
      int q  = tid + e * 256;           // quad index, 512 quads = 2048 halves
      int rr = q >> 3, cc = (q & 7) << 2;
      float4 f = *(const float4*)&emb[(size_t)tok[rr] * HID + k0 + cc];
      v4h h4;
      h4[0] = (_Float16)f.x; h4[1] = (_Float16)f.y;
      h4[2] = (_Float16)f.z; h4[3] = (_Float16)f.w;
      *(v4h*)&Ash[rr][cc] = h4;
    }
    __syncthreads();
    v16h a  = load_a(&Ash[0][0], 40, m0, 0, lane);
    v16h b0 = load_b(Wg, HID, k0, n0, lane);        // B frags straight from global (b128)
    v16h b1 = load_b(Wg, HID, k0, n0 + 16, lane);
    acc0 = wmma_f16(a, b0, acc0);
    acc1 = wmma_f16(a, b1, acc1);
    __syncthreads();
  }

  const float* bias = (g == 0) ? bz : (g == 1) ? br : bh;
#pragma unroll
  for (int i = 0; i < 8; ++i) {
    int m   = m0 + i + ((lane >> 4) << 3);
    int row = Mt * 64 + m;
    int nA  = n0 + (lane & 15);
    int nB  = nA + 16;
    Xg[((size_t)g * SEQ * BAT + row) * HID + nA] = (_Float16)(acc0[i] + bias[nA]);
    Xg[((size_t)g * SEQ * BAT + row) * HID + nB] = (_Float16)(acc1[i] + bias[nB]);
  }
}

// ---------------- grid-wide split barrier (agent scope) ------------------------

__device__ __forceinline__ void grid_sync(unsigned* counter, unsigned* gen) {
  __threadfence();
  __syncthreads();
  if (threadIdx.x == 0) {
    unsigned g = __hip_atomic_load(gen, __ATOMIC_RELAXED, __HIP_MEMORY_SCOPE_AGENT);
    unsigned a = __hip_atomic_fetch_add(counter, 1u, __ATOMIC_ACQ_REL, __HIP_MEMORY_SCOPE_AGENT);
    if (a == (unsigned)NBLK - 1u) {
      __hip_atomic_store(counter, 0u, __ATOMIC_RELAXED, __HIP_MEMORY_SCOPE_AGENT);
      __hip_atomic_fetch_add(gen, 1u, __ATOMIC_RELEASE, __HIP_MEMORY_SCOPE_AGENT);
    } else {
      while (__hip_atomic_load(gen, __ATOMIC_ACQUIRE, __HIP_MEMORY_SCOPE_AGENT) == g)
        __builtin_amdgcn_s_sleep(1);
    }
  }
  __syncthreads();
}

// ---------------- kernel 3: persistent recurrent loop (weights LDS-resident) ---

__global__ __launch_bounds__(256, 1) void gru_recur(
    const _Float16* __restrict__ Wrec, const _Float16* __restrict__ Xg,
    _Float16* __restrict__ hg, _Float16* __restrict__ rhg,
    float* __restrict__ hT, unsigned* __restrict__ sync,
    const float* __restrict__ Wfc, const float* __restrict__ bfc,
    float* __restrict__ out)
{
  extern __shared__ _Float16 smem[];
  _Float16* Wsh = smem;                      // [3][NSL][LDW]
  _Float16* hsh = smem + 3 * NSL * LDW;      // [BAT][LDH], reused for h then r*h

  const int tid  = threadIdx.x, lane = tid & 31, wave = tid >> 5;
  const int blk  = blockIdx.x;
  const int nbase = blk * NSL;
  const int m0 = (wave & 3) << 4;            // 4 M tiles
  const int n0 = (wave >> 2) << 4;           // 2 N tiles (relative to slice)

  // One-time: load this block's 3 weight slices into LDS, column(K)-contiguous.
  for (int idx = tid * 8; idx < 3 * NSL * HID; idx += 256 * 8) {
    int g = idx / (NSL * HID);
    int r = idx - g * (NSL * HID);
    int n = r >> 9, k = r & (HID - 1);
    *(v8h*)&Wsh[(g * NSL + n) * LDW + k] = *(const v8h*)&Wrec[((size_t)g * HID + nbase + n) * HID + k];
  }
  __syncthreads();

  unsigned* counter = sync;
  unsigned* gen     = sync + 64;  // separate cache line

  const int nglob = nbase + n0 + (lane & 15);

  for (int t = 0; t < SEQ; ++t) {
    // --- h -> LDS (async DMA into LDS when the toolchain exposes it)
    copy_h_to_lds(hsh, hg, tid);
    wait_async_lds();
    __syncthreads();

    if (t + 1 < SEQ) {  // warm L2 for next step's x-gate slice
      const _Float16* p = &Xg[((size_t)t + 1) * BAT * HID + (size_t)(tid >> 2) * HID + nbase];
      __builtin_prefetch(p, 0, 1);
    }

    // --- z and r gates: acc = h @ Wrec_slice
    v8f accz = {}, accr = {};
#pragma unroll 4
    for (int k0 = 0; k0 < HID; k0 += 32) {
      v16h a  = load_a(hsh, LDH, m0, k0, lane);
      v16h bz = load_b(&Wsh[0 * NSL * LDW], LDW, k0, n0, lane);
      v16h br = load_b(&Wsh[1 * NSL * LDW], LDW, k0, n0, lane);
      accz = wmma_f16(a, bz, accz);
      accr = wmma_f16(a, br, accr);
    }

    float zf[8], hold[8];
#pragma unroll
    for (int i = 0; i < 8; ++i) {
      int m = m0 + i + ((lane >> 4) << 3);
      float xz = (float)Xg[((size_t)(0 * SEQ + t) * BAT + m) * HID + nglob];
      float xr = (float)Xg[((size_t)(1 * SEQ + t) * BAT + m) * HID + nglob];
      float z  = 1.0f / (1.0f + __expf(-(accz[i] + xz)));
      float rv = 1.0f / (1.0f + __expf(-(accr[i] + xr)));
      float hv = (float)hsh[m * LDH + nglob];
      zf[i] = z; hold[i] = hv;
      rhg[m * HID + nglob] = (_Float16)(rv * hv);   // publish r*h slice
    }
    grid_sync(counter, gen);

    // --- r*h -> LDS (reuse hsh)
    copy_h_to_lds(hsh, rhg, tid);
    wait_async_lds();
    __syncthreads();

    // --- candidate: acc = (r*h) @ Wh_slice
    v8f acch = {};
#pragma unroll 4
    for (int k0 = 0; k0 < HID; k0 += 32) {
      v16h a  = load_a(hsh, LDH, m0, k0, lane);
      v16h bh = load_b(&Wsh[2 * NSL * LDW], LDW, k0, n0, lane);
      acch = wmma_f16(a, bh, acch);
    }

#pragma unroll
    for (int i = 0; i < 8; ++i) {
      int m = m0 + i + ((lane >> 4) << 3);
      float xh   = (float)Xg[((size_t)(2 * SEQ + t) * BAT + m) * HID + nglob];
      float hhat = tanhf(acch[i] + xh);
      float hnew = (1.0f - zf[i]) * hold[i] + zf[i] * hhat;
      hg[m * HID + nglob] = (_Float16)hnew;
      if (t == SEQ - 1) hT[m * HID + nglob] = hnew;
    }
    grid_sync(counter, gen);
  }

  // --- final FC: out[b][c] = hT[b] . Wfc[c] + bfc[c]   (64x2, trivial)
  if (blk == 0 && tid < 128) {
    int b = tid >> 1, c = tid & 1;
    float s = bfc[c];
#pragma unroll 8
    for (int k = 0; k < HID; ++k) s += hT[b * HID + k] * Wfc[c * HID + k];
    out[b * 2 + c] = s;
  }
}

// ---------------- host launch ---------------------------------------------------

extern "C" void kernel_launch(void* const* d_in, const int* in_sizes, int n_in,
                              void* d_out, int out_size, void* d_ws, size_t ws_size,
                              hipStream_t stream) {
  const int*   x   = (const int*)d_in[0];
  const float* emb = (const float*)d_in[1];
  const float* Wz  = (const float*)d_in[2];
  const float* bz  = (const float*)d_in[3];
  const float* Wr  = (const float*)d_in[4];
  const float* br  = (const float*)d_in[5];
  const float* Wh  = (const float*)d_in[6];
  const float* bh  = (const float*)d_in[7];
  const float* Wfc = (const float*)d_in[8];
  const float* bfc = (const float*)d_in[9];
  float* out = (float*)d_out;

  char* ws = (char*)d_ws;
  size_t off = 0;
  _Float16* Wrec = (_Float16*)(ws + off); off += (size_t)3 * HID * HID * 2;
  _Float16* Wx   = (_Float16*)(ws + off); off += (size_t)3 * HID * HID * 2;
  _Float16* Xg   = (_Float16*)(ws + off); off += (size_t)3 * SEQ * BAT * HID * 2;
  _Float16* hg   = (_Float16*)(ws + off); off += (size_t)BAT * HID * 2;
  _Float16* rhg  = (_Float16*)(ws + off); off += (size_t)BAT * HID * 2;
  float*    hT   = (float*)   (ws + off); off += (size_t)BAT * HID * 4;
  unsigned* sync = (unsigned*)(ws + off); off += 512;

  (void)hipMemsetAsync(hg,  0, (size_t)BAT * HID * 2, stream);   // h0 = 0
  (void)hipMemsetAsync(rhg, 0, (size_t)BAT * HID * 2, stream);
  (void)hipMemsetAsync((void*)sync, 0, 512, stream);

  // 1) weight split + f32->f16
  gru_prep<<<(3 * HID * 2 * HID + 255) / 256, 256, 0, stream>>>(Wz, Wr, Wh, Wrec, Wx);

  // 2) input-side gate projections for all timesteps (big WMMA GEMM)
  dim3 gx(HID / 64, (SEQ * BAT) / 64, 3);
  gru_xgates<<<gx, 256, 0, stream>>>(x, emb, Wx, bz, br, bh, Xg);

  // 3) persistent recurrent loop, weights in LDS (166,400 B per WGP)
  size_t smem = (size_t)(3 * NSL * LDW + BAT * LDH) * sizeof(_Float16);
  (void)hipFuncSetAttribute((const void*)gru_recur,
                            hipFuncAttributeMaxDynamicSharedMemorySize, (int)smem);
  gru_recur<<<NBLK, 256, smem, stream>>>(Wrec, Xg, hg, rhg, hT, sync, Wfc, bfc, out);
  (void)in_sizes; (void)n_in; (void)out_size; (void)ws_size;
}
